// LongAttention_16784732192768
// MI455X (gfx1250) — compile-verified, matrix operand order checked by reading, not run on hardware
//
#include <hip/hip_runtime.h>
#include <hip/hip_bf16.h>

// ---------------------------------------------------------------------------
// LongAttention (gated linear attention) for MI455X / gfx1250, wave32 + WMMA.
// GEMMs (6 x 4096x2048x2048 = 206 GFLOP) run on v_wmma_f32_16x16x32_bf16 with
// GLOBAL_LOAD_ASYNC_TO_LDS_B128 staging (ASYNCcnt-tracked DMA, no VGPR bounce).
// Elementwise/scan work is bandwidth-bound and fused aggressively.
// ---------------------------------------------------------------------------

typedef __bf16 bf16_t;
typedef bf16_t bf16x16 __attribute__((ext_vector_type(16)));
typedef bf16_t bf16x8  __attribute__((ext_vector_type(8)));
typedef float  floatx8 __attribute__((ext_vector_type(8)));

#define B_   2
#define T_   2048
#define D_   2048
#define H_   16
#define HD_  128
#define NC_  16              // scan chunks along T
#define TC_  (T_ / NC_)      // 128
#define EPS_GN 1e-5f

static constexpr long MT = (long)B_ * T_;   // 4096 rows
static constexpr long XE = MT * (long)D_;   // 8,388,608 elements (B,T,D)

__device__ __forceinline__ float sigmoidf_(float x) { return 1.f / (1.f + __expf(-x)); }

// CDNA5 async global->LDS copy, 16B per lane, tracked by ASYNCcnt.
__device__ __forceinline__ void async_copy_b128(unsigned lds_off, const void* gaddr) {
  asm volatile("global_load_async_to_lds_b128 %0, %1, off"
               :: "v"(lds_off), "v"((unsigned long long)(uintptr_t)gaddr));
}
__device__ __forceinline__ void async_wait_all() {
  asm volatile("s_wait_asynccnt 0x0" ::: "memory");
}

// ---------------------------------------------------------------------------
// 1) causal depthwise conv (K=4) + SiLU, bf16 casts of x and x_conv,
//    and the (B, D, K-1) conv cache tail.
// ---------------------------------------------------------------------------
__global__ __launch_bounds__(256) void conv_prep_kernel(
    const float* __restrict__ x, const float* __restrict__ cw,
    const float* __restrict__ cbias,
    __hip_bfloat16* __restrict__ xb, __hip_bfloat16* __restrict__ cb,
    float* __restrict__ cache_out)
{
  long i = (long)blockIdx.x * blockDim.x + threadIdx.x;
  if (i >= XE) return;
  int c = (int)(i % D_);
  long bt = i / D_;
  int t = (int)(bt % T_);
  int b = (int)(bt / T_);

  float xv = x[i];
  xb[i] = __float2bfloat16(xv);

  float acc = cbias[c];
  const float* w = cw + (long)c * 4;
#pragma unroll
  for (int j = 0; j < 4; ++j) {
    int tt = t - 3 + j;
    if (tt >= 0) acc += x[((long)b * T_ + tt) * D_ + c] * w[j];
  }
  float s = acc * sigmoidf_(acc);           // SiLU
  cb[i] = __float2bfloat16(s);

  if (t >= T_ - 3)
    cache_out[((long)b * D_ + c) * 3 + (t - (T_ - 3))] = xv;
}

// ---------------------------------------------------------------------------
// 2) f32 -> bf16 weight conversion
// ---------------------------------------------------------------------------
__global__ __launch_bounds__(256) void cast_bf16_kernel(
    const float* __restrict__ src, __hip_bfloat16* __restrict__ dst, long n)
{
  long i = (long)blockIdx.x * blockDim.x + threadIdx.x;
  if (i < n) dst[i] = __float2bfloat16(src[i]);
}

// ---------------------------------------------------------------------------
// 3) WMMA GEMM: Y(MxN) = act(A(MxK) @ W(NxK)^T + bias)
//    128x128 block tile, BK=32, 8 waves (4x2), each wave 2x4 16x16 tiles.
//    Double-buffered LDS filled by GLOBAL_LOAD_ASYNC_TO_LDS_B128; rows padded
//    to 80B so the ds_load_b128 fragment reads are bank-conflict free.
// ---------------------------------------------------------------------------
#define BM 128
#define BN 128
#define BK 32
#define LDT 40   // padded LDS row stride in bf16 elements (80 bytes)

__global__ __launch_bounds__(256) void wmma_gemm_kernel(
    const __hip_bfloat16* __restrict__ A, const __hip_bfloat16* __restrict__ W,
    const float* __restrict__ bias, float* __restrict__ Y,
    int M, int N, int K, int act /*0=none, 1=sigmoid*/)
{
  __shared__ __hip_bfloat16 sA[2][BM * LDT];
  __shared__ __hip_bfloat16 sB[2][BN * LDT];

  const int tid = threadIdx.x;
  const int m0 = blockIdx.y * BM;
  const int n0 = blockIdx.x * BN;

  const int wave = tid >> 5;
  const int lane = tid & 31;
  const int wm = wave & 3;          // 4 waves along M
  const int wn = wave >> 2;         // 2 waves along N
  const int r  = lane & 15;
  const int hi = lane >> 4;

  // Each thread owns 2 of the 512 16B chunks per matrix tile.
  auto load_tile = [&](int stage, int k0) {
#pragma unroll
    for (int i = 0; i < 2; ++i) {
      int lin = tid + i * 256;          // 0..511
      int row = lin >> 2;               // 0..127
      int cc  = (lin & 3) * 8;          // 0,8,16,24 (bf16 units)
      const __hip_bfloat16* ga = A + (size_t)(m0 + row) * K + k0 + cc;
      unsigned la = (unsigned)(uintptr_t)&sA[stage][row * LDT + cc];
      async_copy_b128(la, ga);
      const __hip_bfloat16* gw = W + (size_t)(n0 + row) * K + k0 + cc;
      unsigned lb = (unsigned)(uintptr_t)&sB[stage][row * LDT + cc];
      async_copy_b128(lb, gw);
    }
  };

  floatx8 acc[2][4] = {};

  load_tile(0, 0);
  async_wait_all();
  __syncthreads();

  const int nK = K / BK;
  for (int kt = 0; kt < nK; ++kt) {
    int cur = kt & 1;
    if (kt + 1 < nK) load_tile(cur ^ 1, (kt + 1) * BK);   // async DMA overlaps WMMA

    // A fragments: 16-bit A 16x32 layout — lane r = row, hi selects K-halves.
    bf16x16 af[2];
#pragma unroll
    for (int mi = 0; mi < 2; ++mi) {
      int row = wm * 32 + mi * 16 + r;
      const __hip_bfloat16* p = &sA[cur][row * LDT + hi * 8];
      bf16x8 lo = *(const bf16x8*)p;
      bf16x8 hh = *(const bf16x8*)(p + 16);
      af[mi] = __builtin_shufflevector(lo, hh, 0,1,2,3,4,5,6,7,8,9,10,11,12,13,14,15);
    }
    // B fragments (K x 16, sourced from W rows -> same addressing, lane = n)
    bf16x16 bfr[4];
#pragma unroll
    for (int ni = 0; ni < 4; ++ni) {
      int row = wn * 64 + ni * 16 + r;
      const __hip_bfloat16* p = &sB[cur][row * LDT + hi * 8];
      bf16x8 lo = *(const bf16x8*)p;
      bf16x8 hh = *(const bf16x8*)(p + 16);
      bfr[ni] = __builtin_shufflevector(lo, hh, 0,1,2,3,4,5,6,7,8,9,10,11,12,13,14,15);
    }
#pragma unroll
    for (int mi = 0; mi < 2; ++mi)
#pragma unroll
      for (int ni = 0; ni < 4; ++ni)
        acc[mi][ni] = __builtin_amdgcn_wmma_f32_16x16x32_bf16(
            false, af[mi], false, bfr[ni], (short)0, acc[mi][ni], false, false);

    async_wait_all();     // stage nxt fully landed in LDS
    __syncthreads();      // publish to all waves
  }

  // Epilogue. C layout: lanes 0-15 -> N=lane, M=v; lanes 16-31 -> M=8+v.
#pragma unroll
  for (int mi = 0; mi < 2; ++mi) {
#pragma unroll
    for (int ni = 0; ni < 4; ++ni) {
      int col = n0 + wn * 64 + ni * 16 + r;
      float bv = bias ? bias[col] : 0.f;
#pragma unroll
      for (int v = 0; v < 8; ++v) {
        int rowg = m0 + wm * 32 + mi * 16 + hi * 8 + v;
        float val = acc[mi][ni][v] + bv;
        if (act == 1) val = sigmoidf_(val);
        Y[(size_t)rowg * N + col] = val;
      }
    }
  }
}

// ---------------------------------------------------------------------------
// 4) gamma = sigmoid(x_conv @ Wg^T + b_g): tiny N=16 GEMM, one block per (b,t)
// ---------------------------------------------------------------------------
__global__ __launch_bounds__(256) void gamma_kernel(
    const __hip_bfloat16* __restrict__ cb, const float* __restrict__ Wg,
    const float* __restrict__ bg, float* __restrict__ gamma)
{
  int bt = blockIdx.x;                      // 0..MT-1
  int wave = threadIdx.x >> 5, lane = threadIdx.x & 31;
  const __hip_bfloat16* xr = cb + (size_t)bt * D_;
#pragma unroll
  for (int hh = 0; hh < 2; ++hh) {
    int h = wave * 2 + hh;
    const float* wr = Wg + (long)h * D_;
    float s = 0.f;
    for (int i = lane; i < D_; i += 32) s += __bfloat162float(xr[i]) * wr[i];
#pragma unroll
    for (int off = 16; off; off >>= 1) s += __shfl_xor(s, off, 32);
    if (lane == 0) gamma[(long)bt * H_ + h] = sigmoidf_(s + bg[h]);
  }
}

// ---------------------------------------------------------------------------
// 5) per-(b,t,h): l2norm q,k (in place), bterm = ig*k_n*v (into ig buffer).
//    One wave per head instance; lane holds 4 contiguous channels.
// ---------------------------------------------------------------------------
__global__ __launch_bounds__(256) void prep_qkv_kernel(
    float* __restrict__ q, float* __restrict__ k,
    const float* __restrict__ v, float* __restrict__ ig)
{
  long gw = ((long)blockIdx.x * blockDim.x + threadIdx.x) >> 5; // (b*T+t)*H + h
  if (gw >= MT * H_) return;
  int lane = threadIdx.x & 31;
  size_t base = (size_t)gw * HD_ + lane * 4;

  float4 qv = *(const float4*)&q[base];
  float4 kv = *(const float4*)&k[base];
  float sq = qv.x*qv.x + qv.y*qv.y + qv.z*qv.z + qv.w*qv.w;
  float sk = kv.x*kv.x + kv.y*kv.y + kv.z*kv.z + kv.w*kv.w;
#pragma unroll
  for (int off = 16; off; off >>= 1) {
    sq += __shfl_xor(sq, off, 32);
    sk += __shfl_xor(sk, off, 32);
  }
  float rq = 1.f / fmaxf(sqrtf(sq), 1e-12f);
  float rk = 1.f / fmaxf(sqrtf(sk), 1e-12f);
  qv.x *= rq; qv.y *= rq; qv.z *= rq; qv.w *= rq;
  kv.x *= rk; kv.y *= rk; kv.z *= rk; kv.w *= rk;
  *(float4*)&q[base] = qv;

  float4 vv = *(const float4*)&v[base];
  float4 gv = *(const float4*)&ig[base];
  gv.x *= kv.x * vv.x; gv.y *= kv.y * vv.y;
  gv.z *= kv.z * vv.z; gv.w *= kv.w * vv.w;
  *(float4*)&ig[base] = gv;                 // ig buffer now holds bterm
}

// ---------------------------------------------------------------------------
// 6) chunked scan, pass 1: per (b,h,chunk) compute gamma product + local end
// ---------------------------------------------------------------------------
__global__ __launch_bounds__(128) void scan_pass1_kernel(
    const float* __restrict__ bterm, const float* __restrict__ gamma,
    float* __restrict__ P, float* __restrict__ S)
{
  int blk = blockIdx.x;                  // (b*H + h)*NC + c
  int c = blk % NC_;
  int h = (blk / NC_) % H_;
  int b = blk / (NC_ * H_);
  int d = threadIdx.x;

  float mem = 0.f, p = 1.f;
  for (int tt = 0; tt < TC_; ++tt) {
    int t = c * TC_ + tt;
    float g  = gamma[((long)b * T_ + t) * H_ + h];
    float bt = bterm[((size_t)((long)b * T_ + t) * H_ + h) * HD_ + d];
    mem = g * mem + bt;
    p *= g;
  }
  S[(size_t)blk * HD_ + d] = mem;
  if (d == 0) P[blk] = p;
}

// ---------------------------------------------------------------------------
// 7) carry combine (16 sequential chunk steps per (b,h)); emits rnn state
// ---------------------------------------------------------------------------
__global__ __launch_bounds__(128) void scan_carry_kernel(
    const float* __restrict__ P, const float* __restrict__ S,
    float* __restrict__ Cin, float* __restrict__ out_state)
{
  int bh = blockIdx.x;                   // b*H + h
  int d = threadIdx.x;
  float carry = 0.f;
  for (int c = 0; c < NC_; ++c) {
    int blk = bh * NC_ + c;
    Cin[(size_t)blk * HD_ + d] = carry;
    carry = P[blk] * carry + S[(size_t)blk * HD_ + d];
  }
  out_state[(long)bh * HD_ + d] = carry;  // mem[:, -1] in (B,H,HD) layout
}

// 128-thread reduction of (a,b): shfl within wave, 8-word LDS across 4 waves.
__device__ __forceinline__ float2 block_reduce2_(float a, float b, float* sm) {
#pragma unroll
  for (int off = 16; off; off >>= 1) {
    a += __shfl_xor(a, off, 32);
    b += __shfl_xor(b, off, 32);
  }
  int w = threadIdx.x >> 5, l = threadIdx.x & 31;
  if (l == 0) { sm[w] = a; sm[4 + w] = b; }
  __syncthreads();
  float2 rv;
  rv.x = sm[0] + sm[1] + sm[2] + sm[3];
  rv.y = sm[4] + sm[5] + sm[6] + sm[7];
  __syncthreads();
  return rv;
}

// ---------------------------------------------------------------------------
// 8) scan pass 2 + fused memnorm + q·mem + groupnorm + output gate -> bf16
// ---------------------------------------------------------------------------
__global__ __launch_bounds__(128) void scan_pass2_kernel(
    const float* __restrict__ bterm, const float* __restrict__ gamma,
    const float* __restrict__ qn, const float* __restrict__ og,
    const float* __restrict__ Cin,
    const float* __restrict__ mn_w, const float* __restrict__ mn_b,
    const float* __restrict__ gn_w, const float* __restrict__ gn_b,
    __hip_bfloat16* __restrict__ gated)
{
  __shared__ float sm[8];
  int blk = blockIdx.x;
  int c = blk % NC_;
  int h = (blk / NC_) % H_;
  int b = blk / (NC_ * H_);
  int d = threadIdx.x;
  int ch = h * HD_ + d;

  const float mw = mn_w[ch], mb = mn_b[ch];
  const float gw = gn_w[ch], gb = gn_b[ch];
  const float inv = 1.f / (float)HD_;

  float mem = Cin[(size_t)blk * HD_ + d];

  for (int tt = 0; tt < TC_; ++tt) {
    int t = c * TC_ + tt;
    size_t idx = ((size_t)((long)b * T_ + t) * H_ + h) * HD_ + d;
    float g  = gamma[((long)b * T_ + t) * H_ + h];
    mem = g * mem + bterm[idx];

    float2 s1 = block_reduce2_(mem, mem * mem, sm);
    float mu = s1.x * inv;
    float var = s1.y * inv - mu * mu;
    float mn = (mem - mu) * rsqrtf(var + EPS_GN) * mw + mb;

    float o = mn * qn[idx];
    float2 s2 = block_reduce2_(o, o * o, sm);
    float mu2 = s2.x * inv;
    float var2 = s2.y * inv - mu2 * mu2;
    float gnv = (o - mu2) * rsqrtf(var2 + EPS_GN) * gw + gb;

    gated[idx] = __float2bfloat16(gnv * og[idx]);
  }
}

// ---------------------------------------------------------------------------
// Host-side orchestration
// ---------------------------------------------------------------------------
extern "C" void kernel_launch(void* const* d_in, const int* in_sizes, int n_in,
                              void* d_out, int out_size, void* d_ws, size_t ws_size,
                              hipStream_t stream) {
  const float* x      = (const float*)d_in[0];
  const float* Wq     = (const float*)d_in[1];
  const float* Wk     = (const float*)d_in[2];
  const float* Wv     = (const float*)d_in[3];
  const float* Wo     = (const float*)d_in[4];
  const float* conv_w = (const float*)d_in[5];
  const float* conv_b = (const float*)d_in[6];
  const float* Wig    = (const float*)d_in[7];
  const float* b_ig   = (const float*)d_in[8];
  const float* Wog    = (const float*)d_in[9];
  const float* b_og   = (const float*)d_in[10];
  const float* Wg     = (const float*)d_in[11];
  const float* b_g    = (const float*)d_in[12];
  const float* gn_w   = (const float*)d_in[13];
  const float* gn_b   = (const float*)d_in[14];
  const float* mn_w   = (const float*)d_in[15];
  const float* mn_b   = (const float*)d_in[16];

  float* out_main  = (float*)d_out;                 // (B,T,D)
  float* out_state = out_main + XE;                 // (B,H,HD)
  float* out_cache = out_state + (long)B_ * H_ * HD_; // (B,D,3)

  // workspace carve-up (256B aligned)
  char* ws = (char*)d_ws;
  size_t off = 0;
  auto carve = [&](size_t bytes) {
    size_t o = off; off = (off + bytes + 255) & ~(size_t)255; return o;
  };
  __hip_bfloat16* xb  = (__hip_bfloat16*)(ws + carve(XE * 2));
  __hip_bfloat16* cb  = (__hip_bfloat16*)(ws + carve(XE * 2));
  const size_t WB = (size_t)D_ * D_ * 2;
  __hip_bfloat16* Wqb  = (__hip_bfloat16*)(ws + carve(WB));
  __hip_bfloat16* Wkb  = (__hip_bfloat16*)(ws + carve(WB));
  __hip_bfloat16* Wvb  = (__hip_bfloat16*)(ws + carve(WB));
  __hip_bfloat16* Wigb = (__hip_bfloat16*)(ws + carve(WB));
  __hip_bfloat16* Wogb = (__hip_bfloat16*)(ws + carve(WB));
  __hip_bfloat16* Wob  = (__hip_bfloat16*)(ws + carve(WB));
  float* qb  = (float*)(ws + carve(XE * 4));
  float* kb  = (float*)(ws + carve(XE * 4));
  float* vb  = (float*)(ws + carve(XE * 4));
  float* igb = (float*)(ws + carve(XE * 4));   // becomes bterm
  float* ogb = (float*)(ws + carve(XE * 4));
  float* gam = (float*)(ws + carve(MT * H_ * 4));
  __hip_bfloat16* gated = (__hip_bfloat16*)(ws + carve(XE * 2));
  const int NBLK = B_ * H_ * NC_;              // 512
  float* Pb  = (float*)(ws + carve((size_t)NBLK * 4));
  float* Sb  = (float*)(ws + carve((size_t)NBLK * HD_ * 4));
  float* Cin = (float*)(ws + carve((size_t)NBLK * HD_ * 4));

  // 1) conv + SiLU + bf16 casts + conv cache
  {
    long nb = (XE + 255) / 256;
    conv_prep_kernel<<<dim3((unsigned)nb), dim3(256), 0, stream>>>(
        x, conv_w, conv_b, xb, cb, out_cache);
  }
  // 2) weight bf16 casts
  {
    long n = (long)D_ * D_;
    unsigned nb = (unsigned)((n + 255) / 256);
    cast_bf16_kernel<<<nb, 256, 0, stream>>>(Wq,  Wqb,  n);
    cast_bf16_kernel<<<nb, 256, 0, stream>>>(Wk,  Wkb,  n);
    cast_bf16_kernel<<<nb, 256, 0, stream>>>(Wv,  Wvb,  n);
    cast_bf16_kernel<<<nb, 256, 0, stream>>>(Wig, Wigb, n);
    cast_bf16_kernel<<<nb, 256, 0, stream>>>(Wog, Wogb, n);
    cast_bf16_kernel<<<nb, 256, 0, stream>>>(Wo,  Wob,  n);
  }
  // 3) the five input-side GEMMs (WMMA bf16, f32 accumulate)
  {
    dim3 grid(D_ / BN, (unsigned)(MT / BM));
    wmma_gemm_kernel<<<grid, 256, 0, stream>>>(xb, Wqb,  nullptr, qb,  (int)MT, D_, D_, 0);
    wmma_gemm_kernel<<<grid, 256, 0, stream>>>(xb, Wkb,  nullptr, kb,  (int)MT, D_, D_, 0);
    wmma_gemm_kernel<<<grid, 256, 0, stream>>>(xb, Wvb,  nullptr, vb,  (int)MT, D_, D_, 0);
    wmma_gemm_kernel<<<grid, 256, 0, stream>>>(cb, Wigb, b_ig,    igb, (int)MT, D_, D_, 1);
    wmma_gemm_kernel<<<grid, 256, 0, stream>>>(cb, Wogb, b_og,    ogb, (int)MT, D_, D_, 1);
  }
  // 4) gamma
  gamma_kernel<<<dim3((unsigned)MT), 256, 0, stream>>>(cb, Wg, b_g, gam);
  // 5) l2norm q/k, bterm
  {
    long waves = MT * H_;                       // 65536
    prep_qkv_kernel<<<dim3((unsigned)(waves / 8)), 256, 0, stream>>>(qb, kb, vb, igb);
  }
  // 6-8) chunked scan + fused norms/gating
  scan_pass1_kernel<<<dim3(NBLK), 128, 0, stream>>>(igb, gam, Pb, Sb);
  scan_carry_kernel<<<dim3(B_ * H_), 128, 0, stream>>>(Pb, Sb, Cin, out_state);
  scan_pass2_kernel<<<dim3(NBLK), 128, 0, stream>>>(
      igb, gam, qb, ogb, Cin, mn_w, mn_b, gn_w, gn_b, gated);
  // 9) final projection GEMM
  {
    dim3 grid(D_ / BN, (unsigned)(MT / BM));
    wmma_gemm_kernel<<<grid, 256, 0, stream>>>(gated, Wob, nullptr, out_main,
                                               (int)MT, D_, D_, 0);
  }
}
